// Generator_29978871726096
// MI455X (gfx1250) — matrix operand order repaired and checked by reading, not compiled, not run on hardware
//
#include <hip/hip_runtime.h>
#include <hip/hip_bf16.h>

#define B_ 1024
#define T_ 60
#define V_ 9
#define H_ 512
#define L_ 3
#define Z_ 128

typedef __attribute__((ext_vector_type(16))) __bf16   v16bf;
typedef __attribute__((ext_vector_type(8)))  float    v8f;
typedef __attribute__((ext_vector_type(4)))  unsigned int v4u;

// ---------- scalar helpers ----------
__device__ __forceinline__ unsigned short f2bf(float f) {
  unsigned int u = __builtin_bit_cast(unsigned int, f);
  u += 0x7FFFu + ((u >> 16) & 1u);          // round-to-nearest-even
  return (unsigned short)(u >> 16);
}
__device__ __forceinline__ float sigf(float x) { return 1.0f / (1.0f + __expf(-x)); }

// ---------- WMMA operand loaders (contiguous k = 16*(lane/16) + s in both A and B, wave32) ----------
// A 16x32 bf16 tile from row-major X[M][K]:
//   lane L: m = L&15 ; elements 0..15 = K = k0 + ((L>>4)&1)*16 .. +15
__device__ __forceinline__ v16bf load_tileA(const unsigned short* X, int K, int row0, int k0) {
  int lane = threadIdx.x & 31;
  int m    = lane & 15;
  int koff = (lane >> 4) << 4;              // 0 or 16
  const unsigned short* p = X + (size_t)(row0 + m) * K + k0 + koff;
  union { v4u u[2]; v16bf v; } a;
  a.u[0] = *(const v4u*)(p);                // K = k0+koff    .. +7
  a.u[1] = *(const v4u*)(p + 8);            // K = k0+koff+8  .. +15
  return a.v;
}
// B 32x16 bf16 tile from W^T stored as WT[N][K]:
//   lane L: n = n0 + (L&15) ; elements 0..15 = K = k0 + ((L>>4)&1)*16 .. +15
__device__ __forceinline__ v16bf load_tileB(const unsigned short* WT, int K, int n0, int k0) {
  int lane = threadIdx.x & 31;
  int n    = n0 + (lane & 15);
  int koff = (lane >> 4) << 4;              // 0 or 16
  const unsigned short* p = WT + (size_t)n * K + k0 + koff;
  union { v4u u[2]; v16bf v; } b;
  b.u[0] = *(const v4u*)(p);                // K = k0+koff    .. +7
  b.u[1] = *(const v4u*)(p + 8);            // K = k0+koff+8  .. +15
  return b.v;
}
__device__ __forceinline__ v8f wmma_bf16(v16bf a, v16bf b, v8f c) {
  return __builtin_amdgcn_wmma_f32_16x16x32_bf16(false, a, false, b, (short)0, c, false, false);
}

// ---------- LDS weight-panel staging (pre-swizzled to WMMA lane layout) ----------
// One (gate,kstep) B tile = 32 lanes x 32B = 1KB in LDS; lane slots contiguous.
__device__ __forceinline__ void stage_panel(const unsigned short* __restrict__ WT, int K,
                                            int hc0, int nK, unsigned short* __restrict__ sB,
                                            int waveId) {
  int lane = threadIdx.x & 31;
  for (int ti = waveId; ti < 4 * nK; ti += 8) {   // wave-uniform trip count -> EXEC stays full
    int gate = ti / nK, ks = ti - gate * nK;
    int n    = (gate << 9) + hc0 + (lane & 15);
    int koff = (lane >> 4) << 4;
    const unsigned short* p = WT + (size_t)n * K + ks * 32 + koff;
    v4u lo = *(const v4u*)(p);
    v4u hi = *(const v4u*)(p + 8);
    v4u* d = (v4u*)(sB + (size_t)ti * 512 + lane * 16);
    d[0] = lo;                                    // ds_store_b128 x2
    d[1] = hi;
  }
}
__device__ __forceinline__ v16bf lds_tileB(const unsigned short* __restrict__ sB, int ti) {
  int lane = threadIdx.x & 31;
  const v4u* p = (const v4u*)(sB + (size_t)ti * 512 + lane * 16);
  union { v4u u[2]; v16bf v; } b;
  b.u[0] = p[0];                                  // ds_load_b128 x2
  b.u[1] = p[1];
  return b.v;
}

// ---------- generic WMMA GEMM:  C[M,N] = A[M,K](bf16) * BT[N,K]^T (bf16) + bias ----------
// 2 M-tiles (32 rows) x 1 N-tile per wave; B reused across both M-tiles.
__global__ void gemm_wmma(const unsigned short* __restrict__ A,
                          const unsigned short* __restrict__ BT,
                          const float* __restrict__ bias,
                          float* __restrict__ Cf, unsigned short* __restrict__ Cbf,
                          int M, int N, int K, int leaky) {
  int wave   = (blockIdx.x * blockDim.x + threadIdx.x) >> 5;
  int nTiles = N >> 4;
  int mTile  = wave / nTiles;               // over M/32 row-pairs
  int nTile  = wave % nTiles;
  int m0 = mTile << 5, n0 = nTile << 4;
  int lane = threadIdx.x & 31;
  int nCol = n0 + (lane & 15);

  v8f acc[2];
  float binit = bias ? bias[nCol] : 0.0f;
#pragma unroll
  for (int h = 0; h < 2; ++h)
#pragma unroll
    for (int r = 0; r < 8; ++r) acc[h][r] = binit;

  for (int k0 = 0; k0 < K; k0 += 32) {
    v16bf a0 = load_tileA(A, K, m0,      k0);
    v16bf a1 = load_tileA(A, K, m0 + 16, k0);
    v16bf b  = load_tileB(BT, K, n0, k0);
    acc[0] = wmma_bf16(a0, b, acc[0]);
    acc[1] = wmma_bf16(a1, b, acc[1]);
  }
  int mBase = m0 + ((lane >> 4) << 3);      // C layout: VGPR r -> M = r (+8 for hi lanes)
#pragma unroll
  for (int h = 0; h < 2; ++h)
#pragma unroll
    for (int r = 0; r < 8; ++r) {
      float x = acc[h][r];
      if (leaky) x = (x >= 0.0f) ? x : 0.2f * x;
      size_t idx = (size_t)(mBase + (h << 4) + r) * N + nCol;
      if (Cf)  Cf[idx]  = x;
      if (Cbf) Cbf[idx] = f2bf(x);
    }
}

// ---------- fused LSTM layer ----------
// Block: 8 waves share one 16-col hidden tile (hTile); wave w owns rows mBlock*256+w*32.
// The 4-gate weight panel (64 rows x K) is staged once per block into 64KB LDS
// (pre-swizzled), cutting global weight traffic 8x; Wa and Wb reuse the same buffer.
// Each wave: 8 independent WMMA accumulator chains (2 row-halves x 4 gates).
__global__ void lstm_layer(const unsigned short* __restrict__ Xa,
                           const unsigned short* __restrict__ WaT, int Ka,
                           const unsigned short* __restrict__ Xb,
                           const unsigned short* __restrict__ WbT, int Kb,
                           const float* __restrict__ cinitFull,   // [B,4H] (pre0 incl. biases) or null
                           const float* __restrict__ cinitRow,    // [4H] summed biases or null
                           float* __restrict__ Cst,               // [B,H] f32, in-place
                           unsigned short* __restrict__ Hout) {   // [B,H] bf16
  extern __shared__ unsigned short sB[];    // 4*max(nKa,nKb) KB, <= 64KB
  int waveId = threadIdx.x >> 5;
  int hTile  = blockIdx.x & 31;             // 512/16 hidden col tiles
  int mBlk   = blockIdx.x >> 5;             // 1024/256 row blocks
  int m0  = (mBlk << 8) + (waveId << 5);    // 32 rows per wave
  int hc0 = hTile << 4;
  int lane  = threadIdx.x & 31;
  int col   = hc0 + (lane & 15);
  int mBase = m0 + ((lane >> 4) << 3);

  v8f acc[2][4];
  if (cinitFull) {
#pragma unroll
    for (int h = 0; h < 2; ++h)
#pragma unroll
      for (int g = 0; g < 4; ++g)
#pragma unroll
        for (int r = 0; r < 8; ++r)
          acc[h][g][r] = cinitFull[(size_t)(mBase + (h << 4) + r) * (4 * H_) + (g << 9) + col];
  } else {
#pragma unroll
    for (int g = 0; g < 4; ++g) {
      float b = cinitRow[(g << 9) + col];
#pragma unroll
      for (int r = 0; r < 8; ++r) { acc[0][g][r] = b; acc[1][g][r] = b; }
    }
  }

  // ---- phase A: x-input GEMM ----
  int nKa = Ka >> 5;
  stage_panel(WaT, Ka, hc0, nKa, sB, waveId);
  __syncthreads();
  for (int ks = 0; ks < nKa; ++ks) {
    v16bf a0 = load_tileA(Xa, Ka, m0,      ks << 5);
    v16bf a1 = load_tileA(Xa, Ka, m0 + 16, ks << 5);
#pragma unroll
    for (int g = 0; g < 4; ++g) {
      v16bf b = lds_tileB(sB, g * nKa + ks);
      acc[0][g] = wmma_bf16(a0, b, acc[0][g]);
      acc[1][g] = wmma_bf16(a1, b, acc[1][g]);
    }
  }
  __syncthreads();                          // panel fully consumed before restage

  // ---- phase B: recurrent GEMM ----
  int nKb = Kb >> 5;
  stage_panel(WbT, Kb, hc0, nKb, sB, waveId);
  __syncthreads();
  for (int ks = 0; ks < nKb; ++ks) {
    v16bf a0 = load_tileA(Xb, Kb, m0,      ks << 5);
    v16bf a1 = load_tileA(Xb, Kb, m0 + 16, ks << 5);
#pragma unroll
    for (int g = 0; g < 4; ++g) {
      v16bf b = lds_tileB(sB, g * nKb + ks);
      acc[0][g] = wmma_bf16(a0, b, acc[0][g]);
      acc[1][g] = wmma_bf16(a1, b, acc[1][g]);
    }
  }

  // ---- cell update (torch gate order i,f,g,o), all in registers ----
#pragma unroll
  for (int h = 0; h < 2; ++h)
#pragma unroll
    for (int r = 0; r < 8; ++r) {
      size_t idx = (size_t)(mBase + (h << 4) + r) * H_ + col;
      float gi = sigf(acc[h][0][r]);
      float gf = sigf(acc[h][1][r]);
      float gg = tanhf(acc[h][2][r]);
      float go = sigf(acc[h][3][r]);
      float cn = gf * Cst[idx] + gi * gg;
      Cst[idx]  = cn;
      Hout[idx] = f2bf(go * tanhf(cn));
    }
}

// ---------- prep kernels ----------
__global__ void transpose_bf16(const float* __restrict__ W, int ldw, int Kvalid, int Nvalid,
                               unsigned short* __restrict__ WT, int Kpad, int Npad) {
  int tid = blockIdx.x * blockDim.x + threadIdx.x;
  if (tid >= Kpad * Npad) return;
  int n = tid / Kpad, k = tid % Kpad;
  float v = (k < Kvalid && n < Nvalid) ? W[(size_t)k * ldw + n] : 0.0f;
  WT[(size_t)n * Kpad + k] = f2bf(v);
}

__global__ void bias_sum(const float* a, const float* b, float* o, int n) {
  int t = blockIdx.x * blockDim.x + threadIdx.x;
  if (t < n) o[t] = a[t] + b[t];
}
__global__ void pad_bias(const float* s, float* d, int n, int npad) {
  int t = blockIdx.x * blockDim.x + threadIdx.x;
  if (t < npad) d[t] = (t < n) ? s[t] : 0.0f;
}

__global__ void build_inputs(const float* __restrict__ z, const int* __restrict__ scen,
                             const float* __restrict__ embed,
                             unsigned short* __restrict__ zc, unsigned short* __restrict__ embbf,
                             unsigned short* __restrict__ xext) {
  int tid = blockIdx.x * blockDim.x + threadIdx.x;
  if (tid >= B_ * (Z_ + H_)) return;
  int b = tid / (Z_ + H_), j = tid % (Z_ + H_);
  float v;
  if (j < Z_) v = z[(size_t)b * Z_ + j];
  else {
    v = embed[(size_t)scen[b] * H_ + (j - Z_)];
    embbf[(size_t)b * H_ + (j - Z_)] = f2bf(v);
  }
  zc[(size_t)b * (Z_ + H_) + j] = f2bf(v);
  if (j < 32) xext[b * 32 + j] = 0;   // prev_out at t=0 is zero; cols 9..31 stay zero-pad
}

__global__ void split_init(const float* __restrict__ hc,
                           unsigned short* __restrict__ hbuf, float* __restrict__ cbuf) {
  int tid = blockIdx.x * blockDim.x + threadIdx.x;
  if (tid >= B_ * L_ * H_) return;
  int b = tid / (L_ * H_);
  int r = tid % (L_ * H_);
  int l = r / H_, h = r % H_;
  const size_t BH = (size_t)B_ * H_;
  hbuf[(size_t)l * 2 * BH + (size_t)b * H_ + h] = f2bf(hc[(size_t)b * 3072 + l * 1024 + h]);
  cbuf[(size_t)l * BH     + (size_t)b * H_ + h] =       hc[(size_t)b * 3072 + l * 1024 + 512 + h];
}

__global__ void out_step(const float* __restrict__ res, const float* __restrict__ trend,
                         float* __restrict__ out, unsigned short* __restrict__ xext, int t) {
  int tid = blockIdx.x * blockDim.x + threadIdx.x;
  if (tid >= B_ * V_) return;
  int b = tid / V_, v = tid % V_;
  float val = trend[(size_t)b * 544 + t * V_ + v] + res[b * 16 + v];
  out[((size_t)b * T_ + t) * V_ + v] = val;
  xext[b * 32 + v] = f2bf(val);
}

// ---------- host launcher ----------
extern "C" void kernel_launch(void* const* d_in, const int* in_sizes, int n_in,
                              void* d_out, int out_size, void* d_ws, size_t ws_size,
                              hipStream_t stream) {
  (void)in_sizes; (void)n_in; (void)out_size; (void)ws_size;
  const float* z     = (const float*)d_in[0];
  const int*   scen  = (const int*)d_in[1];
  const float* embed = (const float*)d_in[2];
  const float* fcW   = (const float*)d_in[3];
  const float* fcb   = (const float*)d_in[4];
  const float* Wih[3] = {(const float*)d_in[5], (const float*)d_in[9],  (const float*)d_in[13]};
  const float* Whh[3] = {(const float*)d_in[6], (const float*)d_in[10], (const float*)d_in[14]};
  const float* bih[3] = {(const float*)d_in[7], (const float*)d_in[11], (const float*)d_in[15]};
  const float* bhh[3] = {(const float*)d_in[8], (const float*)d_in[12], (const float*)d_in[16]};
  const float* o1W = (const float*)d_in[17];
  const float* o1b = (const float*)d_in[18];
  const float* o2W = (const float*)d_in[19];
  const float* o2b = (const float*)d_in[20];
  const float* t1W = (const float*)d_in[21];
  const float* t1b = (const float*)d_in[22];
  const float* t2W = (const float*)d_in[23];
  const float* t2b = (const float*)d_in[24];
  float* out = (float*)d_out;

  const size_t BH = (size_t)B_ * H_;
  char* ws = (char*)d_ws;
  size_t off = 0;
  auto alloc = [&](size_t bytes) -> void* {
    void* p = ws + off;
    off = (off + bytes + 255) & ~(size_t)255;
    return p;
  };
  typedef unsigned short us;
  us* W0extT = (us*)alloc((size_t)2048 * 32 * 2);      // layer0 prev_out part, K padded 9->32
  us* WembT  = (us*)alloc((size_t)2048 * 512 * 2);     // layer0 emb part (hoisted)
  us* WhhT[3]; us* WihT[3] = {nullptr, nullptr, nullptr};
  for (int l = 0; l < 3; ++l) WhhT[l] = (us*)alloc((size_t)2048 * 512 * 2);
  WihT[1] = (us*)alloc((size_t)2048 * 512 * 2);
  WihT[2] = (us*)alloc((size_t)2048 * 512 * 2);
  us* fcT   = (us*)alloc((size_t)3072 * 640 * 2);
  us* t1T   = (us*)alloc((size_t)512 * 640 * 2);
  us* t2T   = (us*)alloc((size_t)544 * 512 * 2);
  us* o1T   = (us*)alloc((size_t)512 * 512 * 2);
  us* o2T   = (us*)alloc((size_t)16 * 512 * 2);
  us* zc_bf = (us*)alloc((size_t)B_ * 640 * 2);
  us* em_bf = (us*)alloc(BH * 2);
  us* xext  = (us*)alloc((size_t)B_ * 32 * 2);
  us* hbuf  = (us*)alloc((size_t)L_ * 2 * BH * 2);     // [layer][parity][B*H] bf16
  us* u_bf  = (us*)alloc(BH * 2);                      // lrelu(h2@out1) and temp ztr
  float* pre0   = (float*)alloc(BH * 4 * 4);           // [B][4H] hoisted layer0 gates
  float* hcinit = (float*)alloc((size_t)B_ * 3072 * 4);
  float* trendp = (float*)alloc((size_t)B_ * 544 * 4);
  float* cbuf   = (float*)alloc((size_t)L_ * BH * 4);
  float* resf   = (float*)alloc((size_t)B_ * 16 * 4);
  float* bsum[3]; for (int l = 0; l < 3; ++l) bsum[l] = (float*)alloc(2048 * 4);
  float* t2bp = (float*)alloc(544 * 4);
  float* o2bp = (float*)alloc(16 * 4);

  auto tgrid = [](int total) { return dim3((total + 255) / 256); };
  auto gemm = [&](const us* A, const us* BT, const float* bias, float* Cf, us* Cbf,
                  int M, int N, int K, int leaky) {
    int blocks = (M / 32) * (N / 16) / 8;   // 8 waves per block, 32x16 tile per wave
    gemm_wmma<<<dim3(blocks), dim3(256), 0, stream>>>(A, BT, bias, Cf, Cbf, M, N, K, leaky);
  };
  const size_t LSTM_LDS = 4 * 16 * 1024;    // 64KB weight panel (max of Wa/Wb phases)

  // ---- weight prep (bf16 transpose, zero-padded) ----
  transpose_bf16<<<tgrid(2048 * 32),  256, 0, stream>>>(Wih[0],            2048,   9, 2048, W0extT,  32, 2048);
  transpose_bf16<<<tgrid(2048 * 512), 256, 0, stream>>>(Wih[0] + 9 * 2048, 2048, 512, 2048, WembT,  512, 2048);
  for (int l = 0; l < 3; ++l)
    transpose_bf16<<<tgrid(2048 * 512), 256, 0, stream>>>(Whh[l], 2048, 512, 2048, WhhT[l], 512, 2048);
  transpose_bf16<<<tgrid(2048 * 512), 256, 0, stream>>>(Wih[1], 2048, 512, 2048, WihT[1], 512, 2048);
  transpose_bf16<<<tgrid(2048 * 512), 256, 0, stream>>>(Wih[2], 2048, 512, 2048, WihT[2], 512, 2048);
  transpose_bf16<<<tgrid(3072 * 640), 256, 0, stream>>>(fcW, 3072, 640, 3072, fcT, 640, 3072);
  transpose_bf16<<<tgrid(512 * 640),  256, 0, stream>>>(t1W,  512, 640,  512, t1T, 640,  512);
  transpose_bf16<<<tgrid(544 * 512),  256, 0, stream>>>(t2W,  540, 512,  540, t2T, 512,  544);
  transpose_bf16<<<tgrid(512 * 512),  256, 0, stream>>>(o1W,  512, 512,  512, o1T, 512,  512);
  transpose_bf16<<<tgrid(16 * 512),   256, 0, stream>>>(o2W,    9, 512,    9, o2T, 512,   16);
  for (int l = 0; l < 3; ++l)
    bias_sum<<<tgrid(2048), 256, 0, stream>>>(bih[l], bhh[l], bsum[l], 2048);
  pad_bias<<<tgrid(544), 256, 0, stream>>>(t2b, t2bp, 540, 544);
  pad_bias<<<tgrid(16),  256, 0, stream>>>(o2b, o2bp,   9,  16);

  // ---- batch prep ----
  build_inputs<<<tgrid(B_ * 640), 256, 0, stream>>>(z, scen, embed, zc_bf, em_bf, xext);
  gemm(zc_bf, fcT, fcb, hcinit, nullptr, B_, 3072, 640, 0);            // h_init
  split_init<<<tgrid(B_ * L_ * H_), 256, 0, stream>>>(hcinit, hbuf, cbuf);
  gemm(zc_bf, t1T, t1b, nullptr, u_bf, B_, 512, 640, 1);               // lrelu(z_cond@tr1+b)
  gemm(u_bf, t2T, t2bp, trendp, nullptr, B_, 544, 512, 0);             // trend (N 540->544)
  gemm(em_bf, WembT, bsum[0], pre0, nullptr, B_, 2048, 512, 0);        // hoisted layer0 gates

  // ---- recurrence: 60 steps, h double-buffered ----
  for (int t = 0; t < T_; ++t) {
    int pi = t & 1, po = pi ^ 1;
    us* h0i = hbuf + (0 * 2 + pi) * BH; us* h0o = hbuf + (0 * 2 + po) * BH;
    us* h1i = hbuf + (1 * 2 + pi) * BH; us* h1o = hbuf + (1 * 2 + po) * BH;
    us* h2i = hbuf + (2 * 2 + pi) * BH; us* h2o = hbuf + (2 * 2 + po) * BH;
    lstm_layer<<<128, 256, LSTM_LDS, stream>>>(xext, W0extT, 32,  h0i, WhhT[0], 512,
                                               pre0, nullptr, cbuf,          h0o);
    lstm_layer<<<128, 256, LSTM_LDS, stream>>>(h0o, WihT[1], 512, h1i, WhhT[1], 512,
                                               nullptr, bsum[1], cbuf + BH,  h1o);
    lstm_layer<<<128, 256, LSTM_LDS, stream>>>(h1o, WihT[2], 512, h2i, WhhT[2], 512,
                                               nullptr, bsum[2], cbuf + 2 * BH, h2o);
    gemm(h2o, o1T, o1b, nullptr, u_bf, B_, 512, 512, 1);               // lrelu(h2@out1+b)
    gemm(u_bf, o2T, o2bp, resf, nullptr, B_, 16, 512, 0);              // residual (N -> 16)
    out_step<<<tgrid(B_ * V_), 256, 0, stream>>>(resf, trendp, out, xext, t);
  }
}
